// FPCAWindowAttention_8160437862934
// MI455X (gfx1250) — compile-verified, matrix operand order checked by reading, not run on hardware
//
#include <hip/hip_runtime.h>

// ---------------------------------------------------------------------------
// FPCA window attention, fused per-window kernel for gfx1250 (MI455X).
// One workgroup (128 threads = 4 waves) per window; wave w handles head w.
// All GEMMs via v_wmma_f32_16x16x32_bf16. Softmax in registers via shfl_xor.
// ---------------------------------------------------------------------------

typedef __attribute__((ext_vector_type(16))) __bf16 v16bf;
typedef __attribute__((ext_vector_type(8)))  __bf16 v8bf;
typedef __attribute__((ext_vector_type(4)))  __bf16 v4bf;
typedef __attribute__((ext_vector_type(8)))  float  v8f;

#define LOGIT_MAX 4.6051701859880914f   // log(100)

#define NWIN   4096      // B*nH*nW
#define LQ     64
#define LV     64
#define CDIM   128
#define HEADS  4
#define DHEAD  32

union FragBF { v16bf v; v8bf h[2]; };

// Load a 16x32 bf16 A/B-style fragment from a row-major buffer.
// Lane L<16 : row row0+L,    K = {k0+0..7, k0+16..23}
// Lane L>=16: row row0+L-16, K = {k0+8..15, k0+24..31}
__device__ __forceinline__ v16bf load_frag(const __bf16* buf, int row0, int k0,
                                           int stride, int lane) {
  const int r  = lane & 15;
  const int hi = lane >> 4;
  const __bf16* p = buf + (row0 + r) * stride + k0 + hi * 8;
  FragBF f;
  f.h[0] = *(const v8bf*)(p);
  f.h[1] = *(const v8bf*)(p + 16);
  return f.v;
}

// D tile (16x16 f32 accum, VGPR r: lanes0-15 -> (M=r,N=lane), lanes16-31 -> (M=8+r,N=lane-16))
__device__ __forceinline__ void store_tile_bf16(__bf16* buf, int m0, int n0,
                                                int stride, v8f acc, float bias,
                                                int lane) {
  const int n  = n0 + (lane & 15);
  const int hi = lane >> 4;
#pragma unroll
  for (int r = 0; r < 8; ++r)
    buf[(m0 + r + 8 * hi) * stride + n] = (__bf16)(acc[r] + bias);
}

// Store D tile transposed: bufT[n][m], 8 contiguous bf16 per lane (one b128 DS store)
__device__ __forceinline__ void store_tile_bf16_T(__bf16* bufT, int m0, int n0,
                                                  int strideT, v8f acc, float bias,
                                                  int lane) {
  const int n  = n0 + (lane & 15);
  const int hi = lane >> 4;
  v8bf pk;
#pragma unroll
  for (int r = 0; r < 8; ++r) pk[r] = (__bf16)(acc[r] + bias);
  *(v8bf*)(bufT + n * strideT + m0 + 8 * hi) = pk;
}

__device__ __forceinline__ void store_tile_f32g(float* buf, int m0, int n0,
                                                int stride, v8f acc, float bias,
                                                int lane) {
  const int n  = n0 + (lane & 15);
  const int hi = lane >> 4;
#pragma unroll
  for (int r = 0; r < 8; ++r)
    buf[(size_t)(m0 + r + 8 * hi) * stride + n] = acc[r] + bias;
}

// Stage 64x128 f32 tile (global) -> bf16 (LDS), 128 threads, float4 loads.
__device__ __forceinline__ void stage_tile(const float* __restrict__ src,
                                           __bf16* dst, int tid) {
#pragma unroll
  for (int i = tid; i < (LQ * CDIM / 4); i += 128) {
    float4 f = ((const float4*)src)[i];
    v4bf p = { (__bf16)f.x, (__bf16)f.y, (__bf16)f.z, (__bf16)f.w };
    *(v4bf*)(dst + i * 4) = p;
  }
}

// 64x128 = (64x128) * W^T (+bias). Each wave: 2 N-tiles x 4 M-tiles x 4 K-steps.
template <bool TRANSPOSE>
__device__ __forceinline__ void proj_gemm(const __bf16* in_lds,
                                          const __bf16* __restrict__ Wbf,
                                          const float* __restrict__ bias,
                                          __bf16* dst, int wave, int lane) {
#pragma unroll
  for (int nn = 0; nn < 2; ++nn) {
    const int nt = wave * 2 + nn;
    const int n0 = nt * 16;
    v16bf bfrag[4];
#pragma unroll
    for (int kt = 0; kt < 4; ++kt)
      bfrag[kt] = load_frag(Wbf, n0, kt * 32, CDIM, lane);
    const float bval = bias ? bias[n0 + (lane & 15)] : 0.0f;
#pragma unroll
    for (int mt = 0; mt < 4; ++mt) {
      v8f acc = {};
#pragma unroll
      for (int kt = 0; kt < 4; ++kt) {
        v16bf afrag = load_frag(in_lds, mt * 16, kt * 32, CDIM, lane);
        acc = __builtin_amdgcn_wmma_f32_16x16x32_bf16(
            false, afrag, false, bfrag[kt], (short)0, acc, false, false);
      }
      if (TRANSPOSE)
        store_tile_bf16_T(dst, mt * 16, n0, LV, acc, bval, lane);
      else
        store_tile_bf16(dst, mt * 16, n0, CDIM, acc, bval, lane);
    }
  }
}

__global__ void __launch_bounds__(128)
prep_weights_kernel(const float* __restrict__ Wq, const float* __restrict__ Wk,
                    const float* __restrict__ Wv, const float* __restrict__ Wp,
                    __bf16* __restrict__ ws) {
  const int i = blockIdx.x * blockDim.x + threadIdx.x;   // 0 .. 65535
  const float* src = (i < 16384) ? Wq : (i < 32768) ? Wk : (i < 49152) ? Wv : Wp;
  ws[i] = (__bf16)src[i & 16383];
}

__global__ void __launch_bounds__(128)
fpca_window_attention_kernel(const float* __restrict__ q,
                             const float* __restrict__ k,
                             const float* __restrict__ v,
                             const float* __restrict__ pos,
                             const float* __restrict__ bq,
                             const float* __restrict__ bv,
                             const float* __restrict__ bp,
                             const float* __restrict__ lsc,
                             const __bf16* __restrict__ wbf,
                             float* __restrict__ out_x,
                             float* __restrict__ out_attn) {
  __shared__ __align__(16) unsigned char smem[65536];
  __bf16* in_bf = (__bf16*)(smem);            // 16 KB (staging, later P[0..1])
  __bf16* qp    = (__bf16*)(smem + 16384);    // 16 KB (later P[2..3])
  __bf16* kp    = (__bf16*)(smem + 32768);    // 16 KB (later xh)
  __bf16* vpT   = (__bf16*)(smem + 49152);    // 16 KB, [128][64] = v^T
  __bf16* Pl    = (__bf16*)(smem);            // 32 KB, [4][64][64] bf16
  __bf16* xh    = (__bf16*)(smem + 32768);    // 16 KB, [64][128]

  const int tid  = threadIdx.x;
  const int lane = tid & 31;
  const int wave = tid >> 5;           // == head index
  const int w    = blockIdx.x;         // window id in [0, 4096)
  const int bidx = w >> 8;             // batch index (nH*nW = 256)

  const __bf16* Wq_bf = wbf;
  const __bf16* Wk_bf = wbf + 16384;
  const __bf16* Wv_bf = wbf + 32768;
  const __bf16* Wp_bf = wbf + 49152;

  const size_t tile_off = (size_t)w * (LQ * CDIM);

  // ---- Q projection ----
  stage_tile(q + tile_off, in_bf, tid);
  __syncthreads();
  proj_gemm<false>(in_bf, Wq_bf, bq, qp, wave, lane);
  __syncthreads();

  // ---- K projection (no bias) ----
  stage_tile(k + tile_off, in_bf, tid);
  __syncthreads();
  proj_gemm<false>(in_bf, Wk_bf, nullptr, kp, wave, lane);
  __syncthreads();

  // ---- V projection, stored transposed [c][l] ----
  stage_tile(v + tile_off, in_bf, tid);
  __syncthreads();
  proj_gemm<true>(in_bf, Wv_bf, bv, vpT, wave, lane);
  __syncthreads();

  // ---- Attention scores S = q_h k_h^T (per wave = per head) ----
  const int h = wave;
  v8f S[4][4];
  {
    v16bf bk[4];
#pragma unroll
    for (int nt = 0; nt < 4; ++nt)
      bk[nt] = load_frag(kp, nt * 16, h * DHEAD, CDIM, lane);
#pragma unroll
    for (int mt = 0; mt < 4; ++mt) {
      v16bf aq = load_frag(qp, mt * 16, h * DHEAD, CDIM, lane);
#pragma unroll
      for (int nt = 0; nt < 4; ++nt) {
        v8f z = {};
        S[mt][nt] = __builtin_amdgcn_wmma_f32_16x16x32_bf16(
            false, aq, false, bk[nt], (short)0, z, false, false);
      }
    }
  }
  __syncthreads();  // everyone done reading qp/kp before P overwrites them

  // ---- scale + position bias + softmax (registers + shfl within 16-lane half)
  {
    const float scale = __expf(fminf(lsc[h], LOGIT_MAX));
    const float* posh = pos + (((size_t)bidx * HEADS + h) * LQ) * LV;
    float* attn_g = out_attn + (((size_t)w * HEADS + h) * LQ) * LV;
    __bf16* Ph = Pl + h * (LQ * LV);
    const int hi = lane >> 4, nl = lane & 15;
#pragma unroll
    for (int mt = 0; mt < 4; ++mt) {
#pragma unroll
      for (int r = 0; r < 8; ++r) {
        const int m = mt * 16 + r + 8 * hi;
        float val[4];
#pragma unroll
        for (int nt = 0; nt < 4; ++nt)
          val[nt] = S[mt][nt][r] * scale + posh[m * LV + nt * 16 + nl];
        float mx = fmaxf(fmaxf(val[0], val[1]), fmaxf(val[2], val[3]));
#pragma unroll
        for (int off = 1; off < 16; off <<= 1)
          mx = fmaxf(mx, __shfl_xor(mx, off, 32));
        float e[4], sum = 0.0f;
#pragma unroll
        for (int nt = 0; nt < 4; ++nt) { e[nt] = __expf(val[nt] - mx); sum += e[nt]; }
#pragma unroll
        for (int off = 1; off < 16; off <<= 1)
          sum += __shfl_xor(sum, off, 32);
        const float inv = 1.0f / sum;
#pragma unroll
        for (int nt = 0; nt < 4; ++nt) {
          const float p = e[nt] * inv;
          attn_g[m * LV + nt * 16 + nl] = p;          // f32 output #2
          Ph[m * LV + nt * 16 + nl] = (__bf16)p;      // bf16 for P@V
        }
      }
    }
  }
  __syncthreads();

  // ---- O_h = P_h @ V_h  (M=64, N=32, K=64) ----
  {
    const __bf16* Ph = Pl + h * (LQ * LV);
    v16bf bvf[2][2];
#pragma unroll
    for (int nt2 = 0; nt2 < 2; ++nt2)
#pragma unroll
      for (int kt = 0; kt < 2; ++kt)
        bvf[nt2][kt] = load_frag(vpT, h * DHEAD + nt2 * 16, kt * 32, LV, lane);
#pragma unroll
    for (int mt = 0; mt < 4; ++mt) {
      v16bf ap[2];
#pragma unroll
      for (int kt = 0; kt < 2; ++kt)
        ap[kt] = load_frag(Ph, mt * 16, kt * 32, LV, lane);
#pragma unroll
      for (int nt2 = 0; nt2 < 2; ++nt2) {
        v8f acc = {};
#pragma unroll
        for (int kt = 0; kt < 2; ++kt)
          acc = __builtin_amdgcn_wmma_f32_16x16x32_bf16(
              false, ap[kt], false, bvf[nt2][kt], (short)0, acc, false, false);
        store_tile_bf16(xh, mt * 16, h * DHEAD + nt2 * 16, CDIM, acc, 0.0f, lane);
      }
    }
  }
  __syncthreads();

  // ---- x = concat(O) @ Wp^T + bp  -> global f32 ----
  {
    float* xg = out_x + tile_off;
#pragma unroll
    for (int nn = 0; nn < 2; ++nn) {
      const int nt = wave * 2 + nn;
      const int n0 = nt * 16;
      v16bf bw[4];
#pragma unroll
      for (int kt = 0; kt < 4; ++kt)
        bw[kt] = load_frag(Wp_bf, n0, kt * 32, CDIM, lane);
      const float bval = bp[n0 + (lane & 15)];
#pragma unroll
      for (int mt = 0; mt < 4; ++mt) {
        v8f acc = {};
#pragma unroll
        for (int kt = 0; kt < 4; ++kt) {
          v16bf af = load_frag(xh, mt * 16, kt * 32, CDIM, lane);
          acc = __builtin_amdgcn_wmma_f32_16x16x32_bf16(
              false, af, false, bw[kt], (short)0, acc, false, false);
        }
        store_tile_f32g(xg, mt * 16, n0, CDIM, acc, bval, lane);
      }
    }
  }
}

extern "C" void kernel_launch(void* const* d_in, const int* in_sizes, int n_in,
                              void* d_out, int out_size, void* d_ws, size_t ws_size,
                              hipStream_t stream) {
  (void)in_sizes; (void)n_in; (void)out_size; (void)ws_size;
  const float* q   = (const float*)d_in[0];
  const float* k   = (const float*)d_in[1];
  const float* v   = (const float*)d_in[2];
  const float* pos = (const float*)d_in[3];
  const float* Wq  = (const float*)d_in[4];
  const float* bq  = (const float*)d_in[5];
  const float* Wk  = (const float*)d_in[6];
  const float* Wv  = (const float*)d_in[7];
  const float* bv  = (const float*)d_in[8];
  const float* Wp  = (const float*)d_in[9];
  const float* bp  = (const float*)d_in[10];
  const float* ls  = (const float*)d_in[11];

  float* out_x    = (float*)d_out;
  float* out_attn = out_x + (size_t)NWIN * LQ * CDIM;   // x first, then attn

  __bf16* wbf = (__bf16*)d_ws;                          // 4 x 128x128 bf16

  prep_weights_kernel<<<512, 128, 0, stream>>>(Wq, Wk, Wv, Wp, wbf);
  fpca_window_attention_kernel<<<NWIN, 128, 0, stream>>>(
      q, k, v, pos, bq, bv, bp, ls, wbf, out_x, out_attn);
}
